// qS5SSM_60928406061237
// MI455X (gfx1250) — matrix operand order repaired and checked by reading, not compiled, not run on hardware
//
#include <hip/hip_runtime.h>

// ---------------- problem constants ----------------
#define B_    8
#define L_    4096
#define H_    512
#define P_    512
#define M_    (B_ * L_)          // 32768 rows of (b,l)
#define N1_   (2 * P_)           // 1024: [re | im] columns of Bu / xs
#define CHUNK 256
#define NCH   (L_ / CHUNK)       // 16 chunks per sequence

typedef __attribute__((ext_vector_type(16))) __bf16 v16bf;
typedef __attribute__((ext_vector_type(8)))  float  v8f;

// ---------------- helpers ----------------
static __device__ __forceinline__ unsigned short f2bf(float f) {
    unsigned u = __float_as_uint(f);
    u += 0x7FFFu + ((u >> 16) & 1u);      // round-to-nearest-even
    return (unsigned short)(u >> 16);
}
static __device__ __forceinline__ float bf2f(unsigned short h) {
    return __uint_as_float(((unsigned)h) << 16);
}
// fragment-major swizzle for B-matrix (KxN, row-major logical), tiles K=32 x N=16.
// lane n(0..15) holds col n, K=0..15 (elems 0..15); lanes 16..31 hold K=16..31.
static __device__ __forceinline__ size_t bswz(int k, int n, int K) {
    int kt = k >> 5, ko = k & 31, nt = n >> 4, nr = n & 15;
    int lane = ((ko >> 4) << 4) | nr;
    int i = ko & 15;
    return ((size_t)(nt * (K >> 5) + kt) * 32 + lane) * 16 + i;
}

// ---------------- kernel 1: discretization params ----------------
// lamtab[p*6 + {0..5}] = lam_re, lam_im, lam^CHUNK_re, lam^CHUNK_im, coef_re, coef_im
__global__ void k_params(const float* __restrict__ Lre, const float* __restrict__ Lim,
                         const float* __restrict__ log_step, float* __restrict__ lamtab) {
    int p = blockIdx.x * blockDim.x + threadIdx.x;
    if (p >= P_) return;
    float lr = Lre[p], li = Lim[p];
    float dt = __expf(log_step[p]);
    float er = __expf(lr * dt), ang = li * dt;
    float lam_re = er * __cosf(ang), lam_im = er * __sinf(ang);
    float erc = __expf((float)CHUNK * lr * dt);
    float angc = (float)CHUNK * li * dt;
    float lp_re = erc * __cosf(angc), lp_im = erc * __sinf(angc);
    // coeff = (lam - 1) / Lambda   (complex divide)
    float cr = lam_re - 1.0f, ci = lam_im;
    float den = 1.0f / (lr * lr + li * li);
    float co_re = (cr * lr + ci * li) * den;
    float co_im = (ci * lr - cr * li) * den;
    lamtab[p * 6 + 0] = lam_re; lamtab[p * 6 + 1] = lam_im;
    lamtab[p * 6 + 2] = lp_re;  lamtab[p * 6 + 3] = lp_im;
    lamtab[p * 6 + 4] = co_re;  lamtab[p * 6 + 5] = co_im;
}

// ---------------- kernel 2: u fp32 -> bf16 row-major (M x H), 4 elems/thread ----------------
__global__ void k_cvtU(const float* __restrict__ u, unsigned short* __restrict__ uB) {
    size_t id = ((size_t)blockIdx.x * blockDim.x + threadIdx.x) * 4;
    if (id >= (size_t)M_ * H_) return;
    float4 f = *(const float4*)(u + id);
    ushort4 o;
    o.x = f2bf(f.x); o.y = f2bf(f.y); o.z = f2bf(f.z); o.w = f2bf(f.w);
    *(ushort4*)(uB + id) = o;
}

// ---------------- kernel 3: build swizzled B1 = [B_bar_re | B_bar_im]^T  (K=H x N=2P) ----------------
__global__ void k_buildB1(const float* __restrict__ Bre, const float* __restrict__ Bim,
                          const float* __restrict__ lamtab, unsigned short* __restrict__ B1) {
    size_t id = (size_t)blockIdx.x * blockDim.x + threadIdx.x;
    if (id >= (size_t)P_ * H_) return;
    int p = (int)(id / H_), h = (int)(id % H_);
    float co_re = lamtab[p * 6 + 4], co_im = lamtab[p * 6 + 5];
    float br = Bre[id], bi = Bim[id];
    float bb_re = co_re * br - co_im * bi;
    float bb_im = co_re * bi + co_im * br;
    B1[bswz(h, p,      H_)] = f2bf(bb_re);
    B1[bswz(h, P_ + p, H_)] = f2bf(bb_im);
}

// ---------------- kernel 4: build swizzled B2 (K=2P x N=H): [2*C_re^T ; -2*C_im^T] ----------------
__global__ void k_buildB2(const float* __restrict__ Cre, const float* __restrict__ Cim,
                          unsigned short* __restrict__ B2) {
    size_t id = (size_t)blockIdx.x * blockDim.x + threadIdx.x;
    if (id >= (size_t)N1_ * H_) return;
    int k = (int)(id / H_), n = (int)(id % H_);
    float v = (k < P_) ? (2.0f * Cre[(size_t)n * P_ + k])
                       : (-2.0f * Cim[(size_t)n * P_ + (k - P_)]);
    B2[bswz(k, n, N1_)] = f2bf(v);
}

// ---------------- WMMA GEMM: each wave computes a 64x64 tile (4x4 fragments) ----------------
// A: row-major bf16 (M x K).  Bm: fragment-swizzled bf16 (K x N).  Out: (M x N).
template <int K, int N, bool OUT_BF16>
__global__ __launch_bounds__(256) void gemm_wmma(const unsigned short* __restrict__ A,
                                                 const unsigned short* __restrict__ Bm,
                                                 void* __restrict__ Out) {
    const int lane = threadIdx.x & 31;
    const int wid  = blockIdx.x * (blockDim.x >> 5) + (threadIdx.x >> 5);
    const int WN = N >> 6;                 // 64-wide n tiles
    const int wm = wid / WN, wn = wid % WN;
    const int m0 = wm << 6, n0 = wn << 6;
    const int r = lane & 15, g = lane >> 4;
    const int Kt = K >> 5;
    const int nt0 = n0 >> 4;

    v8f acc[4][4];
#pragma unroll
    for (int i = 0; i < 4; ++i)
#pragma unroll
        for (int j = 0; j < 4; ++j) acc[i][j] = (v8f){};

    union AF { v16bf v; uint4 q[2]; };
    AF a[4], bfr[4];

    // A-fragment k order used here (contiguous, same as bswz): lane group g covers K {16g..16g+15}
    const unsigned short* arow[4];
#pragma unroll
    for (int i = 0; i < 4; ++i)
        arow[i] = A + (size_t)(m0 + 16 * i + r) * K + g * 16;

    for (int kt = 0; kt < Kt; ++kt) {
        const int k0 = kt << 5;
#pragma unroll
        for (int i = 0; i < 4; ++i) {
            a[i].q[0] = *(const uint4*)(arow[i] + k0);
            a[i].q[1] = *(const uint4*)(arow[i] + k0 + 8);
        }
#pragma unroll
        for (int j = 0; j < 4; ++j) {
            size_t bo = ((size_t)((nt0 + j) * Kt + kt) * 32 + lane) * 16;
            bfr[j].q[0] = *(const uint4*)(Bm + bo);
            bfr[j].q[1] = *(const uint4*)(Bm + bo + 8);
        }
#pragma unroll
        for (int i = 0; i < 4; ++i)
#pragma unroll
            for (int j = 0; j < 4; ++j)
                acc[i][j] = __builtin_amdgcn_wmma_f32_16x16x32_bf16(
                    false, a[i].v, false, bfr[j].v, (short)0, acc[i][j], false, false);
    }

    // C/D layout: lane col = lane&15 (+tile n), VGPR rr -> row rr + 8*g (+tile m)
    if (OUT_BF16) {
        unsigned short* O = (unsigned short*)Out;
#pragma unroll
        for (int i = 0; i < 4; ++i)
#pragma unroll
            for (int j = 0; j < 4; ++j)
#pragma unroll
                for (int rr = 0; rr < 8; ++rr) {
                    int row = m0 + 16 * i + rr + 8 * g;
                    O[(size_t)row * N + n0 + 16 * j + r] = f2bf(acc[i][j][rr]);
                }
    } else {
        float* O = (float*)Out;
#pragma unroll
        for (int i = 0; i < 4; ++i)
#pragma unroll
            for (int j = 0; j < 4; ++j)
#pragma unroll
                for (int rr = 0; rr < 8; ++rr) {
                    int row = m0 + 16 * i + rr + 8 * g;
                    O[(size_t)row * N + n0 + 16 * j + r] = acc[i][j][rr];
                }
    }
}

// ---------------- scan phase 1: per-chunk sums  S_c = sum_i lam^(CHUNK-1-i) * Bu_i ----------------
// grid: (P/256, B, NCH), block 256; thread handles one (p, b, chunk)
__global__ __launch_bounds__(256) void k_chunksum(const unsigned short* __restrict__ BuP,
                                                  const float* __restrict__ lamtab,
                                                  float* __restrict__ S) {
    int p = blockIdx.x * blockDim.x + threadIdx.x;
    int b = blockIdx.y, c = blockIdx.z;
    float lr = lamtab[p * 6 + 0], li = lamtab[p * 6 + 1];
    float xr = 0.f, xi = 0.f;
    size_t base = ((size_t)b * L_ + (size_t)c * CHUNK) * N1_;
#pragma unroll 4
    for (int i = 0; i < CHUNK; ++i) {
        float br = bf2f(BuP[base + p]);
        float bi = bf2f(BuP[base + P_ + p]);
        float nr = lr * xr - li * xi + br;
        float ni = lr * xi + li * xr + bi;
        xr = nr; xi = ni;
        base += N1_;
    }
    size_t seq = (size_t)b * P_ + p;
    S[(seq * NCH + c) * 2 + 0] = xr;
    S[(seq * NCH + c) * 2 + 1] = xi;
}

// ---------------- scan phase 2: sequential chain over chunk summaries ----------------
__global__ void k_chain(const float* __restrict__ S, const float* __restrict__ lamtab,
                        float* __restrict__ INIT) {
    int seq = blockIdx.x * blockDim.x + threadIdx.x;   // 0..B*P-1
    if (seq >= B_ * P_) return;
    int p = seq & (P_ - 1);
    float lr = lamtab[p * 6 + 2], li = lamtab[p * 6 + 3];  // lam^CHUNK
    float xr = 0.f, xi = 0.f;
#pragma unroll
    for (int c = 0; c < NCH; ++c) {
        size_t o = ((size_t)seq * NCH + c) * 2;
        INIT[o] = xr; INIT[o + 1] = xi;                 // state entering chunk c
        float sr = S[o], si = S[o + 1];
        float nr = lr * xr - li * xi + sr;
        float ni = lr * xi + li * xr + si;
        xr = nr; xi = ni;
    }
}

// ---------------- scan phase 3: re-scan chunks with correct init; emit xs as bf16 (M x 2P) ----------------
__global__ __launch_bounds__(256) void k_rescan(const unsigned short* __restrict__ BuP,
                                                const float* __restrict__ lamtab,
                                                const float* __restrict__ INIT,
                                                unsigned short* __restrict__ XS) {
    int p = blockIdx.x * blockDim.x + threadIdx.x;
    int b = blockIdx.y, c = blockIdx.z;
    float lr = lamtab[p * 6 + 0], li = lamtab[p * 6 + 1];
    size_t seq = (size_t)b * P_ + p;
    float xr = INIT[(seq * NCH + c) * 2 + 0];
    float xi = INIT[(seq * NCH + c) * 2 + 1];
    size_t base = ((size_t)b * L_ + (size_t)c * CHUNK) * N1_;
#pragma unroll 4
    for (int i = 0; i < CHUNK; ++i) {
        float br = bf2f(BuP[base + p]);
        float bi = bf2f(BuP[base + P_ + p]);
        float nr = lr * xr - li * xi + br;
        float ni = lr * xi + li * xr + bi;
        xr = nr; xi = ni;
        XS[base + p]      = f2bf(xr);
        XS[base + P_ + p] = f2bf(xi);
        base += N1_;
    }
}

// ---------------- launcher ----------------
extern "C" void kernel_launch(void* const* d_in, const int* in_sizes, int n_in,
                              void* d_out, int out_size, void* d_ws, size_t ws_size,
                              hipStream_t stream) {
    const float* Lre      = (const float*)d_in[0];
    const float* Lim      = (const float*)d_in[1];
    const float* Bre      = (const float*)d_in[2];
    const float* Bim      = (const float*)d_in[3];
    const float* Cre      = (const float*)d_in[4];
    const float* Cim      = (const float*)d_in[5];
    const float* log_step = (const float*)d_in[6];
    const float* u        = (const float*)d_in[7];
    float* ys             = (float*)d_out;

    // workspace carve-up (256B aligned)
    uintptr_t w = (uintptr_t)d_ws;
    auto alignup = [](uintptr_t x) { return (x + 255) & ~(uintptr_t)255; };
    float* lamtab = (float*)w;                 w = alignup(w + (size_t)P_ * 6 * 4);
    unsigned short* uB  = (unsigned short*)w;  w = alignup(w + (size_t)M_ * H_ * 2);
    unsigned short* B1  = (unsigned short*)w;  w = alignup(w + (size_t)H_ * N1_ * 2);
    unsigned short* B2  = (unsigned short*)w;  w = alignup(w + (size_t)N1_ * H_ * 2);
    unsigned short* BuP = (unsigned short*)w;  w = alignup(w + (size_t)M_ * N1_ * 2);
    unsigned short* XS  = (unsigned short*)w;  w = alignup(w + (size_t)M_ * N1_ * 2);
    float* S    = (float*)w;                   w = alignup(w + (size_t)B_ * P_ * NCH * 2 * 4);
    float* INIT = (float*)w;                   w = alignup(w + (size_t)B_ * P_ * NCH * 2 * 4);
    (void)ws_size; (void)in_sizes; (void)n_in; (void)out_size;

    // 1) discretization params
    k_params<<<(P_ + 255) / 256, 256, 0, stream>>>(Lre, Lim, log_step, lamtab);
    // 2) u -> bf16 (vectorized x4)
    k_cvtU<<<(int)(((size_t)M_ * H_ / 4 + 255) / 256), 256, 0, stream>>>(u, uB);
    // 3) B1 = discretized B (re|im), swizzled
    k_buildB1<<<(int)(((size_t)P_ * H_ + 255) / 256), 256, 0, stream>>>(Bre, Bim, lamtab, B1);
    // 4) B2 = [2*C_re^T ; -2*C_im^T], swizzled
    k_buildB2<<<(int)(((size_t)N1_ * H_ + 255) / 256), 256, 0, stream>>>(Cre, Cim, B2);
    // 5) GEMM1: Bu(M x 2P) = uB(M x H) @ B1(H x 2P)   [bf16 WMMA, f32 acc, bf16 out]
    {
        int waves = (M_ / 64) * (N1_ / 64);
        gemm_wmma<H_, N1_, true><<<waves / 8, 256, 0, stream>>>(uB, B1, (void*)BuP);
    }
    // 6-8) chunked complex scan over L
    k_chunksum<<<dim3(P_ / 256, B_, NCH), 256, 0, stream>>>(BuP, lamtab, S);
    k_chain<<<(B_ * P_) / 256, 256, 0, stream>>>(S, lamtab, INIT);
    k_rescan<<<dim3(P_ / 256, B_, NCH), 256, 0, stream>>>(BuP, lamtab, INIT, XS);
    // 9) GEMM2: ys(M x H) = XS(M x 2P) @ B2(2P x H)   [f32 out, factor 2 & sign folded]
    {
        int waves = (M_ / 64) * (H_ / 64);
        gemm_wmma<N1_, H_, false><<<waves / 8, 256, 0, stream>>>(XS, B2, (void*)ys);
    }
}
